// GPT_65575560675696
// MI455X (gfx1250) — matrix operand order repaired and checked by reading, not compiled, not run on hardware
//
#include <hip/hip_runtime.h>
#include <hip/hip_bf16.h>

#define B_ 4
#define S_ 1024
#define D_ 1024
#define H_ 16
#define HD_ 64
#define F_ 4096
#define L_ 4
#define V_ 32000
#define M_ (B_ * S_)   // 4096 activation rows

typedef __attribute__((ext_vector_type(16))) __bf16 v16bf;
typedef __attribute__((ext_vector_type(8)))  float  v8f;
typedef __attribute__((ext_vector_type(4)))  unsigned int u32x4;
typedef __attribute__((ext_vector_type(8)))  int i32x8;
typedef __attribute__((ext_vector_type(4)))  int i32x4;

union Frag { v16bf v; unsigned short u[16]; unsigned int w[8]; };
union Acc  { v8f   v; float f[8]; };

#ifndef __has_builtin
#define __has_builtin(x) 0
#endif
#if defined(__AMDGCN__) && __has_builtin(__builtin_amdgcn_tensor_load_to_lds)
#define USE_TDM 1
#else
#define USE_TDM 0
#endif

static __device__ __forceinline__ unsigned short f2bf(float f) {
    unsigned int u = __float_as_uint(f);
    u = (u + 0x7FFFu + ((u >> 16) & 1u)) >> 16;   // round-to-nearest-even
    return (unsigned short)u;
}

static __device__ __forceinline__ v8f wmma_bf16(v16bf a, v16bf b, v8f c) {
    return __builtin_amdgcn_wmma_f32_16x16x32_bf16(false, a, false, b, (short)0, c, false, false);
}

#if USE_TDM
// 2D tile DMA: global (row-major, row_stride elements) -> contiguous LDS.
// data_size_code: 0=1B,1=2B,2=4B,3=8B.  Issued per-wave (EXEC ignored).
static __device__ __forceinline__ void tdm_load_2d(
    unsigned lds_off, const void* gptr, unsigned tile_x, unsigned tile_y,
    unsigned long long row_stride, unsigned data_size_code)
{
    unsigned long long ga = (unsigned long long)gptr;
    u32x4 g0;
    g0.x = 1u;                                              // count=1, user mode
    g0.y = lds_off;                                         // lds_addr [63:32]
    g0.z = (unsigned)ga;                                    // global_addr [95:64]
    g0.w = (unsigned)((ga >> 32) & 0x1FFFFFFu) | (2u << 30);// addr hi + type=2
    i32x8 g1;
    g1[0] = (int)(data_size_code << 16);                    // wg_mask=0, data_size
    g1[1] = (int)(tile_x << 16);                            // tensor_dim0[15:0]
    g1[2] = (int)((tile_x >> 16) | (tile_y << 16));         // dim0 hi | tensor_dim1 lo
    g1[3] = (int)(((tile_y >> 16) & 0xFFFFu) | (tile_x << 16)); // dim1 hi | tile_dim0
    g1[4] = (int)(tile_y & 0xFFFFu);                        // tile_dim1 (tile_dim2=0)
    g1[5] = (int)(unsigned)(row_stride & 0xFFFFFFFFull);    // tensor_dim0_stride lo
    g1[6] = (int)(unsigned)((row_stride >> 32) & 0xFFFFull);// stride hi (dim1_stride=0)
    g1[7] = 0;
    i32x4 z4 = {0, 0, 0, 0};
#if __clang_major__ >= 23
    i32x8 z8 = {0, 0, 0, 0, 0, 0, 0, 0};
    __builtin_amdgcn_tensor_load_to_lds(g0, g1, z4, z4, z8, 0);
#else
    __builtin_amdgcn_tensor_load_to_lds(g0, g1, z4, z4, 0);
#endif
}
#endif

// ---------------------------------------------------------------------------
// Embedding + sqrt(D) + sinusoidal positional encoding -> x (f32) and xb (bf16)
// ---------------------------------------------------------------------------
__global__ __launch_bounds__(256) void embed_kernel(
    const int* __restrict__ tok, const float* __restrict__ emb,
    float* __restrict__ X, unsigned short* __restrict__ Xb)
{
    int row = blockIdx.x;            // b*S + s
    int s   = row & (S_ - 1);
    int t   = threadIdx.x;
    int token = tok[row];
    size_t rb = (size_t)row * D_;
    size_t eb = (size_t)token * D_;
#pragma unroll
    for (int i = 0; i < 4; ++i) {
        int d = t + i * 256;
        int j = d >> 1;
        float ang = (float)s * __powf(10000.f, -(4.f * (float)j) / (float)D_);
        float pe  = (d & 1) ? __cosf(ang) : __sinf(ang);
        float v   = emb[eb + d] + 32.0f /* sqrt(1024) */ + pe;
        X[rb + d]  = v;
        Xb[rb + d] = f2bf(v);
    }
}

// ---------------------------------------------------------------------------
// GEMM: C[M,N] = act(A[M,K](bf16) * B[K,N](f32->bf16) + bias)
// Tile 128x128x32, 256 threads (8 waves), wave = 32x64 (2x4 WMMA frags).
// A tile staged by the Tensor Data Mover; B tile converted f32->bf16 by VALU.
// ---------------------------------------------------------------------------
template <bool GELU, bool OUTF, bool OUTB>
__global__ __launch_bounds__(256) void gemm_bf16_kernel(
    const unsigned short* __restrict__ A, const float* __restrict__ Bw,
    const float* __restrict__ bias, float* __restrict__ Cf,
    unsigned short* __restrict__ Cb, int N, int K)
{
    __shared__ unsigned short As[128][32];   // [m][k]
    __shared__ unsigned short Bs[128][32];   // [n][k] (transposed)

    int tid  = threadIdx.x;
    int lane = tid & 31, wid = tid >> 5;
    int l16 = lane & 15, hlf = lane >> 4;
    int m0 = blockIdx.y * 128, n0 = blockIdx.x * 128;
    int Moff = (wid >> 1) * 32, Noff = (wid & 1) * 64;

    Acc acc[2][4];
#pragma unroll
    for (int a = 0; a < 2; ++a)
#pragma unroll
        for (int b = 0; b < 4; ++b)
#pragma unroll
            for (int r = 0; r < 8; ++r) acc[a][b].f[r] = 0.f;

    for (int k0 = 0; k0 < K; k0 += 32) {
#if USE_TDM
        if (wid == 0) {   // DMA the A tile: 128 rows x 32 bf16, row stride K
            tdm_load_2d((unsigned)(unsigned long long)&As[0][0],
                        A + (size_t)m0 * K + k0, 32u, 128u,
                        (unsigned long long)K, 1u);
        }
#else
        {
            int m = tid >> 1, kc = (tid & 1) * 16;
            const uint4* src = (const uint4*)(A + (size_t)(m0 + m) * K + k0 + kc);
            uint4* dst = (uint4*)&As[m][kc];
            dst[0] = src[0]; dst[1] = src[1];
        }
#endif
        { // stage B tile (f32 -> bf16, transposed)
            int k = tid >> 3, n8 = (tid & 7) * 16;
            const float* src = Bw + (size_t)(k0 + k) * N + n0 + n8;
#pragma unroll
            for (int i = 0; i < 16; ++i) Bs[n8 + i][k] = f2bf(src[i]);
        }
#if USE_TDM
        if (wid == 0) __builtin_amdgcn_s_wait_tensorcnt(0);
#endif
        __syncthreads();

        Frag af[2], bf[4];
#pragma unroll
        for (int mf = 0; mf < 2; ++mf) {
            const unsigned int* rp = (const unsigned int*)&As[Moff + mf * 16 + l16][0];
#pragma unroll
            for (int j = 0; j < 8; ++j) {
                af[mf].w[j]     = rp[hlf * 8 + j];       // K = hlf*16 + 2j, +1
            }
        }
#pragma unroll
        for (int nf = 0; nf < 4; ++nf) {
            const unsigned int* cp = (const unsigned int*)&Bs[Noff + nf * 16 + l16][0];
#pragma unroll
            for (int j = 0; j < 8; ++j) bf[nf].w[j] = cp[hlf * 8 + j]; // K = hlf*16 + i
        }
#pragma unroll
        for (int mf = 0; mf < 2; ++mf)
#pragma unroll
            for (int nf = 0; nf < 4; ++nf)
                acc[mf][nf].v = wmma_bf16(af[mf].v, bf[nf].v, acc[mf][nf].v);
        __syncthreads();
    }

#pragma unroll
    for (int mf = 0; mf < 2; ++mf)
#pragma unroll
        for (int nf = 0; nf < 4; ++nf) {
            int Ng = n0 + Noff + nf * 16 + l16;
            float bv = bias[Ng];
#pragma unroll
            for (int r = 0; r < 8; ++r) {
                int Mg = m0 + Moff + mf * 16 + r + hlf * 8;
                float v = acc[mf][nf].f[r] + bv;
                if (GELU) v = 0.5f * v * (1.f + erff(v * 0.70710678f)); // exact GELU
                size_t off = (size_t)Mg * N + Ng;
                if (OUTF) Cf[off] = v;
                if (OUTB) Cb[off] = f2bf(v);
            }
        }
}

// ---------------------------------------------------------------------------
// Causal flash attention.  Grid: (S/128, B*H).  8 waves; wave w handles 16
// queries q0 = qbase + 16*w.  K/V 32x64 tiles staged via TDM into LDS.
// ---------------------------------------------------------------------------
__global__ __launch_bounds__(256) void attn_kernel(
    const unsigned short* __restrict__ Q, const unsigned short* __restrict__ Kb,
    const unsigned short* __restrict__ Vb, unsigned short* __restrict__ O)
{
    __shared__ unsigned short Kt[32][64];
    __shared__ unsigned short Vt[32][64];
    __shared__ unsigned short Pl[8][16][32];

    int tid = threadIdx.x, lane = tid & 31, wid = tid >> 5;
    int l16 = lane & 15, hlf = lane >> 4;
    int bh = blockIdx.y, b = bh >> 4, h = bh & 15;
    int qbase = blockIdx.x * 128;
    int q0 = qbase + wid * 16;
    size_t base = (size_t)b * S_ * D_ + (size_t)h * HD_;

    Frag aq[2];
#pragma unroll
    for (int f = 0; f < 2; ++f)
#pragma unroll
        for (int i = 0; i < 16; ++i) {
            int hd = f * 32 + hlf * 16 + i;
            aq[f].u[i] = Q[base + (size_t)(q0 + l16) * D_ + hd];
        }

    Acc o[4];
    float mrow[8], lrow[8];
#pragma unroll
    for (int a = 0; a < 4; ++a)
#pragma unroll
        for (int r = 0; r < 8; ++r) o[a].f[r] = 0.f;
#pragma unroll
    for (int r = 0; r < 8; ++r) { mrow[r] = -1e30f; lrow[r] = 0.f; }

    for (int ks = 0; ks < qbase + 128; ks += 32) {
#if USE_TDM
        if (wid == 0) {   // DMA K and V tiles: 32 rows x 64 bf16, row stride D
            tdm_load_2d((unsigned)(unsigned long long)&Kt[0][0],
                        Kb + base + (size_t)ks * D_, 64u, 32u,
                        (unsigned long long)D_, 1u);
            tdm_load_2d((unsigned)(unsigned long long)&Vt[0][0],
                        Vb + base + (size_t)ks * D_, 64u, 32u,
                        (unsigned long long)D_, 1u);
            __builtin_amdgcn_s_wait_tensorcnt(0);
        }
#else
        {
            int row = tid >> 3, c8 = (tid & 7) * 8;
            *(uint4*)&Kt[row][c8] = *(const uint4*)&Kb[base + (size_t)(ks + row) * D_ + c8];
            *(uint4*)&Vt[row][c8] = *(const uint4*)&Vb[base + (size_t)(ks + row) * D_ + c8];
        }
#endif
        __syncthreads();

        if (ks < q0 + 16) {   // wave-uniform activity test
            Acc sc[2];
#pragma unroll
            for (int n = 0; n < 2; ++n) {
#pragma unroll
                for (int r = 0; r < 8; ++r) sc[n].f[r] = 0.f;
#pragma unroll
                for (int kk = 0; kk < 2; ++kk) {
                    Frag bk;
                    const unsigned int* p =
                        (const unsigned int*)&Kt[n * 16 + l16][kk * 32 + hlf * 16];
#pragma unroll
                    for (int j = 0; j < 8; ++j) bk.w[j] = p[j];
                    sc[n].v = wmma_bf16(aq[kk].v, bk.v, sc[n].v);
                }
            }
            float p0s[8], p1s[8];
#pragma unroll
            for (int r = 0; r < 8; ++r) {
                int qg  = q0 + r + hlf * 8;
                int k0g = ks + l16, k1g = ks + 16 + l16;
                float s0 = sc[0].f[r] * 0.125f;  // 1/sqrt(64)
                float s1 = sc[1].f[r] * 0.125f;
                if (k0g > qg) s0 = -1e30f;
                if (k1g > qg) s1 = -1e30f;
                float mx = fmaxf(s0, s1);
#pragma unroll
                for (int off = 8; off; off >>= 1) mx = fmaxf(mx, __shfl_xor(mx, off, 32));
                float mn = fmaxf(mrow[r], mx);
                float al = __expf(mrow[r] - mn);
                float p0 = (k0g > qg) ? 0.f : __expf(s0 - mn);
                float p1 = (k1g > qg) ? 0.f : __expf(s1 - mn);
                float rs = p0 + p1;
#pragma unroll
                for (int off = 8; off; off >>= 1) rs += __shfl_xor(rs, off, 32);
                lrow[r] = lrow[r] * al + rs;
                mrow[r] = mn;
#pragma unroll
                for (int a = 0; a < 4; ++a) o[a].f[r] *= al;
                p0s[r] = p0; p1s[r] = p1;
            }
            // stage P (C layout) -> LDS -> reload as A fragment (wave-private)
#pragma unroll
            for (int r = 0; r < 8; ++r) {
                Pl[wid][r + hlf * 8][l16]      = f2bf(p0s[r]);
                Pl[wid][r + hlf * 8][16 + l16] = f2bf(p1s[r]);
            }
            asm volatile("s_wait_dscnt 0" ::: "memory");
            Frag pf;
#pragma unroll
            for (int i = 0; i < 16; ++i) {
                int kcol = hlf * 16 + i;
                pf.u[i] = Pl[wid][l16][kcol];
            }
#pragma unroll
            for (int a = 0; a < 4; ++a) {
                Frag bv;
#pragma unroll
                for (int i = 0; i < 16; ++i) bv.u[i] = Vt[hlf * 16 + i][a * 16 + l16];
                o[a].v = wmma_bf16(pf.v, bv.v, o[a].v);
            }
        }
        __syncthreads();
    }

#pragma unroll
    for (int a = 0; a < 4; ++a)
#pragma unroll
        for (int r = 0; r < 8; ++r) {
            int qg = q0 + r + hlf * 8;
            float v = o[a].f[r] / lrow[r];
            O[base + (size_t)qg * D_ + a * 16 + l16] = f2bf(v);
        }
}

// ---------------------------------------------------------------------------
// x = x + layernorm(z)*g + be ; refresh bf16 copy of x.  One row per block.
// ---------------------------------------------------------------------------
__global__ __launch_bounds__(256) void ln_res_kernel(
    const float* __restrict__ Z, float* __restrict__ X,
    unsigned short* __restrict__ Xb,
    const float* __restrict__ g, const float* __restrict__ be)
{
    __shared__ float red[256];
    int row = blockIdx.x, t = threadIdx.x;
    size_t rb = (size_t)row * D_;
    float v[4], s = 0.f;
#pragma unroll
    for (int i = 0; i < 4; ++i) { v[i] = Z[rb + t + i * 256]; s += v[i]; }
    red[t] = s; __syncthreads();
    for (int o = 128; o; o >>= 1) { if (t < o) red[t] += red[t + o]; __syncthreads(); }
    float mu = red[0] * (1.f / D_);
    __syncthreads();
    s = 0.f;
#pragma unroll
    for (int i = 0; i < 4; ++i) { float d = v[i] - mu; s += d * d; }
    red[t] = s; __syncthreads();
    for (int o = 128; o; o >>= 1) { if (t < o) red[t] += red[t + o]; __syncthreads(); }
    float rs = rsqrtf(red[0] * (1.f / D_) + 1e-5f);
#pragma unroll
    for (int i = 0; i < 4; ++i) {
        int d = t + i * 256;
        float xn = X[rb + d] + ((v[i] - mu) * rs * g[d] + be[d]);
        X[rb + d]  = xn;
        Xb[rb + d] = f2bf(xn);
    }
}

// ---------------------------------------------------------------------------
// Row softmax over V=32000, in place on f32 logits.
// ---------------------------------------------------------------------------
__global__ __launch_bounds__(256) void softmax_kernel(float* __restrict__ P)
{
    __shared__ float red[256];
    int row = blockIdx.x, t = threadIdx.x;
    float* p = P + (size_t)row * V_;
    float mx = -1e30f;
    for (int i = t; i < V_; i += 256) mx = fmaxf(mx, p[i]);
    red[t] = mx; __syncthreads();
    for (int o = 128; o; o >>= 1) { if (t < o) red[t] = fmaxf(red[t], red[t + o]); __syncthreads(); }
    mx = red[0]; __syncthreads();
    float s = 0.f;
    for (int i = t; i < V_; i += 256) s += __expf(p[i] - mx);
    red[t] = s; __syncthreads();
    for (int o = 128; o; o >>= 1) { if (t < o) red[t] += red[t + o]; __syncthreads(); }
    float inv = 1.f / red[0];
    for (int i = t; i < V_; i += 256) p[i] = __expf(p[i] - mx) * inv;
}

// ---------------------------------------------------------------------------
extern "C" void kernel_launch(void* const* d_in, const int* in_sizes, int n_in,
                              void* d_out, int out_size, void* d_ws, size_t ws_size,
                              hipStream_t stream) {
    (void)in_sizes; (void)n_in; (void)out_size; (void)ws_size;
    const int*   tok = (const int*)  d_in[0];
    const float* emb = (const float*)d_in[1];
    const float* Wq  = (const float*)d_in[2];   const float* bq = (const float*)d_in[3];
    const float* Wk  = (const float*)d_in[4];   const float* bk = (const float*)d_in[5];
    const float* Wv  = (const float*)d_in[6];   const float* bv = (const float*)d_in[7];
    const float* Wo  = (const float*)d_in[8];   const float* bo = (const float*)d_in[9];
    const float* W1  = (const float*)d_in[10];  const float* b1 = (const float*)d_in[11];
    const float* W2  = (const float*)d_in[12];  const float* b2 = (const float*)d_in[13];
    const float* g1  = (const float*)d_in[14];  const float* be1 = (const float*)d_in[15];
    const float* g2  = (const float*)d_in[16];  const float* be2 = (const float*)d_in[17];
    const float* Wh  = (const float*)d_in[18];  const float* bh = (const float*)d_in[19];
    float* out = (float*)d_out;

    char* w = (char*)d_ws;
    size_t off = 0;
    auto alloc = [&](size_t bytes) { char* p = w + off; off += (bytes + 255) & ~(size_t)255; return p; };
    float*          X   = (float*)         alloc((size_t)M_ * D_ * 4);
    unsigned short* Xb  = (unsigned short*)alloc((size_t)M_ * D_ * 2);
    unsigned short* Qb  = (unsigned short*)alloc((size_t)M_ * D_ * 2);
    unsigned short* Kbuf= (unsigned short*)alloc((size_t)M_ * D_ * 2);
    unsigned short* Vbuf= (unsigned short*)alloc((size_t)M_ * D_ * 2);
    unsigned short* Ctx = (unsigned short*)alloc((size_t)M_ * D_ * 2);
    float*          Zf  = (float*)         alloc((size_t)M_ * D_ * 4);
    unsigned short* Hb  = (unsigned short*)alloc((size_t)M_ * F_ * 2);

    dim3 blk(256);
    embed_kernel<<<M_, blk, 0, stream>>>(tok, emb, X, Xb);

    for (int l = 0; l < L_; ++l) {
        const size_t dd = (size_t)D_ * D_, df = (size_t)D_ * F_;
        dim3 gD(D_ / 128, M_ / 128);     // N=1024
        dim3 gF(F_ / 128, M_ / 128);     // N=4096
        gemm_bf16_kernel<false, false, true><<<gD, blk, 0, stream>>>(
            Xb, Wq + l * dd, bq + l * D_, nullptr, Qb, D_, D_);
        gemm_bf16_kernel<false, false, true><<<gD, blk, 0, stream>>>(
            Xb, Wk + l * dd, bk + l * D_, nullptr, Kbuf, D_, D_);
        gemm_bf16_kernel<false, false, true><<<gD, blk, 0, stream>>>(
            Xb, Wv + l * dd, bv + l * D_, nullptr, Vbuf, D_, D_);
        attn_kernel<<<dim3(S_ / 128, B_ * H_), blk, 0, stream>>>(Qb, Kbuf, Vbuf, Ctx);
        gemm_bf16_kernel<false, true, false><<<gD, blk, 0, stream>>>(
            Ctx, Wo + l * dd, bo + l * D_, Zf, nullptr, D_, D_);
        ln_res_kernel<<<M_, blk, 0, stream>>>(Zf, X, Xb, g1 + l * D_, be1 + l * D_);
        gemm_bf16_kernel<true, false, true><<<gF, blk, 0, stream>>>(
            Xb, W1 + l * df, b1 + l * F_, nullptr, Hb, F_, D_);
        gemm_bf16_kernel<false, true, false><<<gD, blk, 0, stream>>>(
            Hb, W2 + l * df, b2 + l * D_, Zf, nullptr, D_, F_);
        ln_res_kernel<<<M_, blk, 0, stream>>>(Zf, X, Xb, g2 + l * D_, be2 + l * D_);
    }
    gemm_bf16_kernel<false, true, false><<<dim3(V_ / 128, M_ / 128), blk, 0, stream>>>(
        Xb, Wh, bh, out, nullptr, V_, D_);
    softmax_kernel<<<M_, blk, 0, stream>>>(out);
}
